// SelfAttention_91079076479707
// MI455X (gfx1250) — matrix operand order repaired and checked by reading, not compiled, not run on hardware
//
#include <hip/hip_runtime.h>

typedef __bf16 bf16;
typedef __attribute__((ext_vector_type(16))) __bf16 v16bf;
typedef __attribute__((ext_vector_type(8)))  __bf16 v8bf;
typedef __attribute__((ext_vector_type(8)))  float  v8f;
typedef __attribute__((ext_vector_type(4)))  float  v4f;

#define B_  4
#define T_  2048
#define D_  1024
#define DK_ 128

// ---------- helpers ----------

static __device__ __forceinline__ bf16 f2bf(float f) {
  return (bf16)f;                       // native v_cvt bf16 path on gfx1250
}

static __device__ __forceinline__ v16bf cat16(v8bf lo, v8bf hi) {
  return __builtin_shufflevector(lo, hi, 0, 1, 2, 3, 4, 5, 6, 7,
                                         8, 9, 10, 11, 12, 13, 14, 15);
}

static __device__ __forceinline__ v8f cat8f(v4f lo, v4f hi) {
  return __builtin_shufflevector(lo, hi, 0, 1, 2, 3, 4, 5, 6, 7);
}

static __device__ __forceinline__ v8f wmma_bf16(v16bf a, v16bf b, v8f c) {
  // D = A(16x32 bf16) * B(32x16 bf16) + C(16x16 f32)
  return __builtin_amdgcn_wmma_f32_16x16x32_bf16(false, a, false, b, (short)0, c,
                                                 false, false);
}

// A-operand (16x32, 16-bit): lane<16 -> M=lane, K = kc + {0..7, 16..23}
//                            lane>=16 -> M=lane-16, K = kc + {8..15, 24..31}
static __device__ __forceinline__ v16bf load_a_bf16(const bf16* base, int ld,
                                                    int row0, int kc, int lane) {
  int half = lane >> 4, m = lane & 15;
  const bf16* p = base + (size_t)(row0 + m) * ld + kc + half * 8;
  v8bf lo = *(const v8bf*)(p);
  v8bf hi = *(const v8bf*)(p + 16);
  return cat16(lo, hi);
}

// B-operand (32x16, 16-bit) from [N][K] storage: lane&15 -> N column,
// lanes 0-15 hold K = kc+{0..7, 16..23}, lanes 16-31 hold K = kc+{8..15, 24..31}.
static __device__ __forceinline__ v16bf load_b_bf16(const bf16* base, int ld,
                                                    int n0, int kc, int lane) {
  int half = lane >> 4, n = lane & 15;
  const bf16* p = base + (size_t)(n0 + n) * ld + kc + half * 8;
  v8bf lo = *(const v8bf*)(p);
  v8bf hi = *(const v8bf*)(p + 16);
  return cat16(lo, hi);
}

// A-operand from fp32 row-major, converted to bf16 with packed converts.
static __device__ __forceinline__ v16bf load_a_f32(const float* base, int ld,
                                                   int row0, int kc, int lane) {
  int half = lane >> 4, m = lane & 15;
  const float* p = base + (size_t)(row0 + m) * ld + kc + half * 8;
  v8f lo = cat8f(*(const v4f*)(p),      *(const v4f*)(p + 4));
  v8f hi = cat8f(*(const v4f*)(p + 16), *(const v4f*)(p + 20));
  v8bf lobf = __builtin_convertvector(lo, v8bf);
  v8bf hibf = __builtin_convertvector(hi, v8bf);
  return cat16(lobf, hibf);
}

// ---------- kernel 0: weight convert + transpose ----------
__global__ __launch_bounds__(256) void prep_weights_kernel(
    const float* __restrict__ Wq, const float* __restrict__ Wk,
    const float* __restrict__ Wv, const float* __restrict__ Wo,
    bf16* __restrict__ WqT, bf16* __restrict__ WkT,
    bf16* __restrict__ WvT, bf16* __restrict__ WoT) {
  int idx = blockIdx.x * blockDim.x + threadIdx.x;
  if (idx >= D_ * DK_) return;
  int k = idx / DK_, n = idx % DK_;           // Wq/Wk/Wv: [D][DK]
  WqT[(size_t)n * D_ + k] = f2bf(Wq[idx]);
  WkT[(size_t)n * D_ + k] = f2bf(Wk[idx]);
  WvT[(size_t)n * D_ + k] = f2bf(Wv[idx]);
  int kk = idx / D_, nn = idx % D_;           // Wo: [DK][D]
  WoT[(size_t)nn * DK_ + kk] = f2bf(Wo[idx]);
}

// ---------- kernel 1: QKV projections (Q,K row-major bf16; V transposed) ----------
__global__ __launch_bounds__(128) void qkv_kernel(
    const float* __restrict__ x,
    const bf16* __restrict__ WqT, const bf16* __restrict__ WkT,
    const bf16* __restrict__ WvT,
    bf16* __restrict__ Q, bf16* __restrict__ K, bf16* __restrict__ Vt) {
  int lane = threadIdx.x & 31, wid = threadIdx.x >> 5;
  int row0 = (blockIdx.x * 4 + wid) * 16;     // flat row in [0, B*T)
  int mat  = blockIdx.y;
  const bf16* WT = (mat == 0) ? WqT : (mat == 1) ? WkT : WvT;

  v8f acc[8] = {};
  v16bf a_cur = load_a_f32(x, D_, row0, 0, lane);
  for (int kc = 0; kc < D_; kc += 32) {
    // Issue all B-fragment loads for this chunk first, then the WMMAs, so the
    // scheduler can overlap the matrix pipe with outstanding loads.
    v16bf bfr[8];
#pragma unroll
    for (int nt = 0; nt < 8; ++nt)
      bfr[nt] = load_b_bf16(WT, D_, nt * 16, kc, lane);
    v16bf a_next = a_cur;
    if (kc + 32 < D_) a_next = load_a_f32(x, D_, row0, kc + 32, lane);
#pragma unroll
    for (int nt = 0; nt < 8; ++nt)
      acc[nt] = wmma_bf16(a_cur, bfr[nt], acc[nt]);
    a_cur = a_next;
  }
  int half = lane >> 4, nl = lane & 15;
  if (mat < 2) {
    bf16* out = (mat == 0) ? Q : K;
#pragma unroll
    for (int nt = 0; nt < 8; ++nt)
#pragma unroll
      for (int r = 0; r < 8; ++r) {
        int row = row0 + r + half * 8;
        out[(size_t)row * DK_ + nt * 16 + nl] = f2bf(acc[nt][r]);
      }
  } else {
#pragma unroll
    for (int nt = 0; nt < 8; ++nt)
#pragma unroll
      for (int r = 0; r < 8; ++r) {
        int row = row0 + r + half * 8;        // = b*T + t
        int b = row >> 11, t = row & (T_ - 1);
        int dv = nt * 16 + nl;
        Vt[((size_t)b * DK_ + dv) * T_ + t] = f2bf(acc[nt][r]);
      }
  }
}

// ---------- kernel 2: causal flash attention (one wave per 16-query tile) ----------
__global__ __launch_bounds__(128) void attn_kernel(
    const bf16* __restrict__ Q, const bf16* __restrict__ K,
    const bf16* __restrict__ Vt, bf16* __restrict__ O) {
  __shared__ bf16 Plds[4][16 * 32];           // per-wave 16x32 P staging
  int lane = threadIdx.x & 31, wid = threadIdx.x >> 5;
  int half = lane >> 4, nl = lane & 15;
  int qtile = blockIdx.x * 4 + wid;
  int gq0 = qtile * 16;
  int b   = gq0 >> 11;
  int q0  = gq0 & (T_ - 1);
  const bf16* Qb  = Q  + (size_t)b * T_ * DK_;
  const bf16* Kb  = K  + (size_t)b * T_ * DK_;
  const bf16* Vtb = Vt + (size_t)b * DK_ * T_;

  v16bf qa[4];
#pragma unroll
  for (int c = 0; c < 4; ++c) qa[c] = load_a_bf16(Qb, DK_, q0, c * 32, lane);

  v8f of[8] = {};
  float mrow[8], lrow[8];
#pragma unroll
  for (int r = 0; r < 8; ++r) { mrow[r] = -1e30f; lrow[r] = 0.f; }

  const float scale = 0.08838834764831845f;   // 1/sqrt(128)
  bf16* Pw = Plds[wid];
  int nkt = (q0 + 15) / 32 + 1;               // causal key-tile bound
  for (int kt = 0; kt < nkt; ++kt) {
    int kb0 = kt * 32;
    // prefetch next K tile (one row per lane -> global_prefetch_b8)
    if (kt + 1 < nkt)
      __builtin_prefetch(Kb + (size_t)(kb0 + 32 + lane) * DK_, 0, 1);

    // ---- S = Q * K^T : issue all K-fragment loads, then 8 WMMAs ----
    v16bf bk[8];
#pragma unroll
    for (int c = 0; c < 4; ++c) {
      bk[2 * c]     = load_b_bf16(Kb, DK_, kb0,      c * 32, lane);
      bk[2 * c + 1] = load_b_bf16(Kb, DK_, kb0 + 16, c * 32, lane);
    }
    v8f s0 = {}, s1 = {};
#pragma unroll
    for (int c = 0; c < 4; ++c) {
      s0 = wmma_bf16(qa[c], bk[2 * c],     s0);
      s1 = wmma_bf16(qa[c], bk[2 * c + 1], s1);
    }

    // ---- online softmax over 32 keys (rows split across 16-lane halves) ----
#pragma unroll
    for (int r = 0; r < 8; ++r) {
      int m = r + half * 8;
      int qrow = q0 + m;
      int k0 = kb0 + nl, k1 = k0 + 16;
      float v0 = (k0 <= qrow) ? s0[r] * scale : -1e30f;
      float v1 = (k1 <= qrow) ? s1[r] * scale : -1e30f;
      float mx = fmaxf(v0, v1);
#pragma unroll
      for (int off = 1; off < 16; off <<= 1)
        mx = fmaxf(mx, __shfl_xor(mx, off, 32));
      float mnew  = fmaxf(mrow[r], mx);
      float alpha = __expf(mrow[r] - mnew);
      float p0 = __expf(v0 - mnew);
      float p1 = __expf(v1 - mnew);
      float rs = p0 + p1;
#pragma unroll
      for (int off = 1; off < 16; off <<= 1)
        rs += __shfl_xor(rs, off, 32);
      lrow[r] = lrow[r] * alpha + rs;
      mrow[r] = mnew;
#pragma unroll
      for (int dc = 0; dc < 8; ++dc) of[dc][r] = of[dc][r] * alpha;
      Pw[m * 32 + nl]      = f2bf(p0);
      Pw[m * 32 + 16 + nl] = f2bf(p1);
    }
    asm volatile("" ::: "memory");
    __builtin_amdgcn_wave_barrier();          // DS ops in-order within the wave

    // ---- O += P(16x32) @ V-tile : load all V fragments, then 8 WMMAs ----
    {
      const bf16* pp = Pw + nl * 32 + half * 8;
      v8bf plo = *(const v8bf*)(pp);
      v8bf phi = *(const v8bf*)(pp + 16);
      v16bf pa = cat16(plo, phi);
      v16bf bv[8];
#pragma unroll
      for (int dc = 0; dc < 8; ++dc)
        bv[dc] = load_b_bf16(Vtb, T_, dc * 16, kb0, lane);
#pragma unroll
      for (int dc = 0; dc < 8; ++dc)
        of[dc] = wmma_bf16(pa, bv[dc], of[dc]);
    }
    asm volatile("" ::: "memory");
    __builtin_amdgcn_wave_barrier();
  }
  // normalize + store O (bf16)
#pragma unroll
  for (int r = 0; r < 8; ++r) {
    float inv = 1.0f / lrow[r];
    int grow = gq0 + r + half * 8;
#pragma unroll
    for (int dc = 0; dc < 8; ++dc)
      O[(size_t)grow * DK_ + dc * 16 + nl] = f2bf(of[dc][r] * inv);
  }
}

// ---------- kernel 3: output projection O @ Wo -> fp32 ----------
__global__ __launch_bounds__(128) void oproj_kernel(
    const bf16* __restrict__ O, const bf16* __restrict__ WoT,
    float* __restrict__ out) {
  int lane = threadIdx.x & 31, wid = threadIdx.x >> 5;
  int half = lane >> 4, nl = lane & 15;
  int row0 = (blockIdx.x * 4 + wid) * 16;
  int n0   = blockIdx.y * 128;
  v8f acc[8] = {};
#pragma unroll
  for (int c = 0; c < 4; ++c) {
    v16bf a = load_a_bf16(O, DK_, row0, c * 32, lane);
    v16bf bfr[8];
#pragma unroll
    for (int nt = 0; nt < 8; ++nt)
      bfr[nt] = load_b_bf16(WoT, DK_, n0 + nt * 16, c * 32, lane);
#pragma unroll
    for (int nt = 0; nt < 8; ++nt)
      acc[nt] = wmma_bf16(a, bfr[nt], acc[nt]);
  }
#pragma unroll
  for (int nt = 0; nt < 8; ++nt)
#pragma unroll
    for (int r = 0; r < 8; ++r) {
      int row = row0 + r + half * 8;
      out[(size_t)row * D_ + n0 + nt * 16 + nl] = acc[nt][r];
    }
}

// ---------- host launcher ----------
extern "C" void kernel_launch(void* const* d_in, const int* in_sizes, int n_in,
                              void* d_out, int out_size, void* d_ws, size_t ws_size,
                              hipStream_t stream) {
  (void)in_sizes; (void)n_in; (void)out_size; (void)ws_size;
  const float* x  = (const float*)d_in[0];
  const float* Wq = (const float*)d_in[1];
  const float* Wk = (const float*)d_in[2];
  const float* Wv = (const float*)d_in[3];
  const float* Wo = (const float*)d_in[4];
  float* out = (float*)d_out;

  char* ws = (char*)d_ws;
  bf16* WqT = (bf16*)(ws);                  // [128][1024] bf16 : 256 KB
  bf16* WkT = (bf16*)(ws +  262144);
  bf16* WvT = (bf16*)(ws +  524288);
  bf16* WoT = (bf16*)(ws +  786432);        // [1024][128] bf16 : 256 KB
  bf16* Q   = (bf16*)(ws + 1048576);        // [B][T][128] bf16 : 2 MB
  bf16* K   = (bf16*)(ws + 3145728);        // [B][T][128] bf16 : 2 MB
  bf16* Vt  = (bf16*)(ws + 5242880);        // [B][128][T] bf16 : 2 MB
  bf16* O   = (bf16*)(ws + 7340032);        // [B][T][128] bf16 : 2 MB

  prep_weights_kernel<<<dim3((D_ * DK_ + 255) / 256), dim3(256), 0, stream>>>(
      Wq, Wk, Wv, Wo, WqT, WkT, WvT, WoT);
  qkv_kernel<<<dim3(128, 3), dim3(128), 0, stream>>>(x, WqT, WkT, WvT, Q, K, Vt);
  attn_kernel<<<dim3(128), dim3(128), 0, stream>>>(Q, K, Vt, O);
  oproj_kernel<<<dim3(128, 8), dim3(128), 0, stream>>>(O, WoT, out);
}
